// FeatExtractor_8796093022837
// MI455X (gfx1250) — matrix operand order repaired and checked by reading, not compiled, not run on hardware
//
#include <hip/hip_runtime.h>

// CDNA5 (gfx1250) wave32 WMMA types
typedef __attribute__((ext_vector_type(16))) _Float16 v16h;
typedef __attribute__((ext_vector_type(8)))  float    v8f;

// Problem constants (from reference): B=16, CIN=12, IN=128, O=32
// conv1: K1 = 12*9 = 108, padded to 128 (4 chunks of 32), M1 = 64
// conv2: K2 = 64*9 = 576 (18 chunks of 32),                M2 = 128

__global__ void init_out_kernel(float* __restrict__ out) {
  out[blockIdx.x * 256 + threadIdx.x] = 0.0f;
}

// Pack W1 (64,12,3,3) f32 -> f16, layout [o][kk(4)][half(2)][p(16)],
// where element p of (kk,half) is K = kk*32 + half*16 + p
// (contiguous k order, same as the B operand in LDS). K>=108 zero-padded.
__global__ void pack_w1_kernel(const float* __restrict__ W1,
                               _Float16* __restrict__ W1p) {
  int i = blockIdx.x * 256 + threadIdx.x;        // 64*128 = 8192
  int o  = i >> 7;
  int rem = i & 127;
  int kk = rem >> 5, r2 = rem & 31, h = r2 >> 4, p = r2 & 15;
  int k = kk * 32 + h * 16 + p;
  float v = 0.0f;
  if (k < 108) {
    int c = k / 9, r = k % 9, ky = r / 3, kx = r % 3;
    v = W1[((o * 12 + c) * 3 + ky) * 3 + kx];
  }
  W1p[i] = (_Float16)v;
}

// Pack W2 (128,64,3,3) f32 -> f16, layout [o][kk(18)][half(2)][p(16)]
__global__ void pack_w2_kernel(const float* __restrict__ W2,
                               _Float16* __restrict__ W2p) {
  int i = blockIdx.x * 256 + threadIdx.x;        // 128*576 = 73728
  int o  = i / 576;
  int rem = i % 576;
  int kk = rem >> 5, r2 = rem & 31, h = r2 >> 4, p = r2 & 15;
  int k = kk * 32 + h * 16 + p;
  int c = k / 9, r = k % 9, ky = r / 3, kx = r % 3;
  W2p[i] = (_Float16)W2[((o * 64 + c) * 3 + ky) * 3 + kx];
}

// conv1 (12->64, 3x3, pad 1) + bias + ReLU + 2x2 maxpool, fused.
// Block = 256 threads = 8 wave32s. Each block covers 2 rows x 16 cols of
// conv output (one pool row-pair) for all 64 channels.
__global__ void conv1_pool_kernel(const float* __restrict__ img,       // (16,12,128,128)
                                  const _Float16* __restrict__ W1p,    // packed
                                  const float* __restrict__ b1,        // (64)
                                  _Float16* __restrict__ pooled) {     // (16,64,64,64)
  __shared__ __align__(32) _Float16 patches[2][16][128];  // 8 KB, K padded to 128
  __shared__ float convout[64][32];                       // 8 KB (64 ch x 2x16 px)

  const int b  = blockIdx.z;
  const int y0 = blockIdx.y * 2;     // conv rows y0, y0+1
  const int x0 = blockIdx.x * 16;    // conv cols x0..x0+15
  const int tid = threadIdx.x;

  // Zero-fill K pad tail (k = 108..127), 2*16*20 = 640 elements.
  for (int e = tid; e < 640; e += 256) {
    int ry = e / 320, rem = e - ry * 320;
    int n = rem / 20, k = 108 + (rem - n * 20);
    patches[ry][n][k] = (_Float16)0.0f;
  }

  // im2col staging: one thread per (tap r, pixel n); bounds check hoisted,
  // inner loop over channels is pure pointer arithmetic.
  if (tid < 144) {                           // 9 taps x 16 pixels
    const int n  = tid & 15;
    const int r  = tid >> 4;                 // 0..8
    const int ky = r / 3, kx = r - 3 * ky;
    const int ix = x0 + n + kx - 1;
    const bool okx = (ix >= 0) & (ix < 128);
#pragma unroll
    for (int ry = 0; ry < 2; ++ry) {
      const int iy = y0 + ry + ky - 1;
      const bool ok = okx & (iy >= 0) & (iy < 128);
      const float* gp = img + ((b * 12) * 128 + iy) * 128 + ix;  // +c*16384
      _Float16* lp = &patches[ry][n][r];                         // +c*9
#pragma unroll 4
      for (int c = 0; c < 12; ++c) {
        lp[c * 9] = (_Float16)(ok ? gp[c * 16384] : 0.0f);
      }
    }
  }
  __syncthreads();

  const int w    = tid >> 5;
  const int lane = tid & 31;
  const int ry   = w >> 2;          // row within pair
  const int mt   = w & 3;           // channel tile
  const int hf   = lane >> 4;       // lane half
  const int ln   = lane & 15;       // A-row (out ch) / B-col (pixel)

  v8f acc = {};
  const v16h* Ap = (const v16h*)W1p;
#pragma unroll
  for (int kk = 0; kk < 4; ++kk) {
    v16h a  = Ap[((mt * 16 + ln) * 4 + kk) * 2 + hf];                  // 32B load
    v16h bm = *(const v16h*)&patches[ry][ln][kk * 32 + hf * 16];       // 2x ds_b128
    acc = __builtin_amdgcn_wmma_f32_16x16x32_f16(
        false, a, false, bm, (short)0, acc, false, false);
  }

  // bias + ReLU -> LDS (D layout: VGPR r holds M = r + 8*hf, N = ln)
#pragma unroll
  for (int r = 0; r < 8; ++r) {
    int m = mt * 16 + hf * 8 + r;
    convout[m][ry * 16 + ln] = fmaxf(acc[r] + b1[m], 0.0f);
  }
  __syncthreads();

  // 2x2 maxpool across the two rows -> 64 ch x 8 pooled cols, store f16
#pragma unroll
  for (int it = 0; it < 2; ++it) {
    int idx = tid + it * 256;                 // 0..511
    int ch = idx >> 3, px = idx & 7;
    float v = fmaxf(fmaxf(convout[ch][2 * px],      convout[ch][2 * px + 1]),
                    fmaxf(convout[ch][16 + 2 * px], convout[ch][16 + 2 * px + 1]));
    pooled[((b * 64 + ch) * 64 + blockIdx.y) * 64 + (blockIdx.x * 8 + px)] =
        (_Float16)v;
  }
}

// conv2 (64->128, 3x3, pad 1) + bias + ReLU + segment-max, fused.
// Block = 256 threads = 8 wave32s covering 16 pixels x all 128 channels.
// Segment-max: out pre-zeroed; post-ReLU values >= 0, so bitwise u32
// atomicMax == float max (order-independent -> deterministic), and empty
// segments correctly stay 0 (matches maximum(segment_max, 0)).
__global__ void conv2_segmax_kernel(const _Float16* __restrict__ pooled, // (16,64,64,64)
                                    const _Float16* __restrict__ W2p,    // packed
                                    const float* __restrict__ b2,        // (128)
                                    const int* __restrict__ rois,        // (16,128,128)
                                    float* __restrict__ out) {           // (16,32,128)
  __shared__ __align__(32) _Float16 patches[16][576];  // 18 KB

  const int b  = blockIdx.z;
  const int y  = blockIdx.y;         // row in 64x64
  const int x0 = blockIdx.x * 16;    // 16-pixel column tile
  const int tid = threadIdx.x;

  // im2col staging: one thread per (tap r, pixel n); channel loop is pure
  // pointer arithmetic (global stride 64*64, LDS stride 9 halfs).
  if (tid < 144) {
    const int n  = tid & 15;
    const int r  = tid >> 4;                 // 0..8
    const int ky = r / 3, kx = r - 3 * ky;
    const int iy = y + ky - 1;
    const int ix = x0 + n + kx - 1;
    const bool ok = (iy >= 0) & (iy < 64) & (ix >= 0) & (ix < 64);
    const _Float16* gp = pooled + ((b * 64) * 64 + iy) * 64 + ix;  // +c*4096
    _Float16* lp = &patches[n][r];                                 // +c*9
    const _Float16 z = (_Float16)0.0f;
#pragma unroll 8
    for (int c = 0; c < 64; ++c) {
      lp[c * 9] = ok ? gp[c * 4096] : z;
    }
  }
  __syncthreads();

  const int mt   = tid >> 5;        // channel tile 0..7
  const int lane = tid & 31;
  const int hf   = lane >> 4;
  const int ln   = lane & 15;

  v8f acc = {};
  const v16h* Ap = (const v16h*)W2p + ((mt * 16 + ln) * 18) * 2 + hf;  // +2 per kk
#pragma unroll 6
  for (int kk = 0; kk < 18; ++kk) {
    // next A chunk is L2-resident; warm WGP$ (global_prefetch_b8)
    __builtin_prefetch(&Ap[(kk + 1) * 2], 0, 1);
    v16h a  = Ap[kk * 2];
    v16h bm = *(const v16h*)&patches[ln][kk * 32 + hf * 16];
    acc = __builtin_amdgcn_wmma_f32_16x16x32_f16(
        false, a, false, bm, (short)0, acc, false, false);
  }

  // nearest-downsampled ROI label for pixel (y, x)
  const int x = x0 + ln;
  const int label = rois[(b * 128 + 2 * y) * 128 + 2 * x];
  unsigned* segbase = (unsigned*)&out[(b * 32 + label) * 128];

#pragma unroll
  for (int r = 0; r < 8; ++r) {
    int m = mt * 16 + hf * 8 + r;
    float v = fmaxf(acc[r] + b2[m], 0.0f);
    atomicMax(&segbase[m], __float_as_uint(v));   // monotone for v >= 0
  }
}

extern "C" void kernel_launch(void* const* d_in, const int* in_sizes, int n_in,
                              void* d_out, int out_size, void* d_ws, size_t ws_size,
                              hipStream_t stream) {
  const float* images = (const float*)d_in[0];   // (16,4,3,128,128) == (16,12,128,128)
  const float* W1     = (const float*)d_in[1];   // (64,12,3,3)
  const float* b1     = (const float*)d_in[2];   // (64)
  const float* W2     = (const float*)d_in[3];   // (128,64,3,3)
  const float* b2     = (const float*)d_in[4];   // (128)
  const int*   rois   = (const int*)d_in[5];     // (16,128,128)
  float* out = (float*)d_out;                    // (16,32,128)

  char* ws = (char*)d_ws;
  _Float16* W1p    = (_Float16*)(ws);            //  16 KB
  _Float16* W2p    = (_Float16*)(ws + 16384);    // 144 KB
  _Float16* pooled = (_Float16*)(ws + 163840);   //   8 MB  (16,64,64,64) f16

  init_out_kernel<<<256, 256, 0, stream>>>(out);                 // 65536 f32
  pack_w1_kernel<<<32, 256, 0, stream>>>(W1, W1p);               // 8192
  pack_w2_kernel<<<288, 256, 0, stream>>>(W2, W2p);              // 73728
  conv1_pool_kernel<<<dim3(8, 64, 16), 256, 0, stream>>>(images, W1p, b1, pooled);
  conv2_segmax_kernel<<<dim3(4, 64, 16), 256, 0, stream>>>(pooled, W2p, b2, rois, out);
}
